// LlamaAttention_23321672417881
// MI455X (gfx1250) — matrix operand order repaired and checked by reading, not compiled, not run on hardware
//
#include <hip/hip_runtime.h>
#include <hip/hip_bf16.h>

#define S_LEN 2048
#define HID   2048
#define NH    16
#define HD    128

typedef __attribute__((ext_vector_type(16))) __bf16 v16bf;
typedef __attribute__((ext_vector_type(8)))  __bf16 v8bf;
typedef __attribute__((ext_vector_type(8)))  float  v8f;

// ---------------- WMMA fragment helpers (wave32, ISA 7.12.2 layouts) ----------------
// A fragment 16x32 bf16: lane r=lane&15 holds row M=r; half h=lane>>4:
//   VGPR0-7: K = h*16 .. h*16+15 (contiguous, matches load_b_frag)
__device__ __forceinline__ v16bf load_a_frag(const __bf16* base, int ld) {
    const int lane = threadIdx.x & 31;
    const int r    = lane & 15;
    const int h    = lane >> 4;
    const v8bf a0 = *(const v8bf*)(base + r * ld + h * 16);
    const v8bf a1 = *(const v8bf*)(base + r * ld + h * 16 + 8);
    v16bf f;
#pragma unroll
    for (int i = 0; i < 8; ++i) { f[i] = a0[i]; f[8 + i] = a1[i]; }
    return f;
}

// B fragment 32x16 bf16 from K-contiguous storage: element (k,n) at baseT[n*ld + k].
// lanes 0-15 hold column n=lane, K=0..15; lanes 16-31 hold K=16..31.
__device__ __forceinline__ v16bf load_b_frag(const __bf16* baseT, int ld) {
    const int lane = threadIdx.x & 31;
    const int r    = lane & 15;
    const int h    = lane >> 4;
    return *(const v16bf*)(baseT + r * ld + h * 16);
}

__device__ __forceinline__ v8f wmma_bf16(v16bf a, v16bf b, v8f c) {
    return __builtin_amdgcn_wmma_f32_16x16x32_bf16(false, a, false, b, (short)0, c,
                                                   false, false);
}

// Async global -> LDS 128-bit copy (CDNA5, tracked on ASYNCcnt).
// LDS aperture maps flat addr[31:0] to the wave-relative LDS byte address (ISA 10.2),
// so the low 32 bits of the flat pointer are the VDST operand.
__device__ __forceinline__ void async_copy_b128(const __bf16* g, __bf16* l) {
    const unsigned loff = (unsigned)(size_t)l;
    asm volatile("global_load_async_to_lds_b128 %0, %1, off"
                 :: "v"(loff), "v"(g)
                 : "memory");
}

__device__ __forceinline__ void wait_asynccnt0() {
    asm volatile("s_wait_asynccnt 0" ::: "memory");
}

// ---------------- QKV projection: C = X @ W^T, f32 in, bf16 out (+RoPE) ------------
// mode 1: RoPE, write [head][s][d]   (Q, K)
// mode 0: no RoPE, write transposed [head][d][s]   (V)
__global__ __launch_bounds__(256)
void qkv_proj_kernel(const float* __restrict__ X, const float* __restrict__ W,
                     __bf16* __restrict__ dst, int mode) {
    __shared__ __align__(32) __bf16 At[64 * 32];
    __shared__ __align__(32) __bf16 Bt[128 * 32];
    __shared__ __align__(16) float  Ct[64 * 128];

    const int tid  = threadIdx.x;
    const int wave = tid >> 5;
    const int wm   = wave & 3;      // 16-row strip
    const int wn   = wave >> 2;     // 64-col half
    const int lane = tid & 31;
    const int r    = lane & 15;
    const int h    = lane >> 4;
    const int r0   = blockIdx.x * 64;
    const int head = blockIdx.y;
    const int c0   = head * HD;

    v8f acc[4] = {};

    const int am = tid >> 2;        // 0..63
    const int ak = (tid & 3) * 8;   // 0,8,16,24
    const int bn = tid >> 1;        // 0..127
    const int bk = (tid & 1) * 16;  // 0,16

    for (int k0 = 0; k0 < HID; k0 += 32) {
        // stage A tile 64x32 (f32 -> bf16)
        {
            const float4* s = (const float4*)(X + (size_t)(r0 + am) * HID + k0 + ak);
            float4 x0 = s[0], x1 = s[1];
            __bf16* d = At + am * 32 + ak;
            d[0] = (__bf16)x0.x; d[1] = (__bf16)x0.y; d[2] = (__bf16)x0.z; d[3] = (__bf16)x0.w;
            d[4] = (__bf16)x1.x; d[5] = (__bf16)x1.y; d[6] = (__bf16)x1.z; d[7] = (__bf16)x1.w;
        }
        // stage B tile 128x32 as [n][k]; W[o][hid] row-major => contiguous in k
        {
            const float4* s = (const float4*)(W + (size_t)(c0 + bn) * HID + k0 + bk);
            __bf16* d = Bt + bn * 32 + bk;
#pragma unroll
            for (int q = 0; q < 4; ++q) {
                float4 x = s[q];
                d[q * 4 + 0] = (__bf16)x.x; d[q * 4 + 1] = (__bf16)x.y;
                d[q * 4 + 2] = (__bf16)x.z; d[q * 4 + 3] = (__bf16)x.w;
            }
        }
        if (k0 + 32 < HID) {
            __builtin_prefetch(X + (size_t)(r0 + am) * HID + k0 + 32 + ak, 0, 1);
            __builtin_prefetch(W + (size_t)(c0 + bn) * HID + k0 + 32 + bk, 0, 1);
        }
        __syncthreads();

        v16bf a = load_a_frag(At + (wm * 16) * 32, 32);
#pragma unroll
        for (int t = 0; t < 4; ++t) {
            v16bf b = load_b_frag(Bt + (wn * 64 + t * 16) * 32, 32);
            acc[t] = wmma_bf16(a, b, acc[t]);
        }
        __syncthreads();
    }

    // spill C fragments to f32 LDS tile so the RoPE pair (c, c^64) is visible
#pragma unroll
    for (int t = 0; t < 4; ++t) {
#pragma unroll
        for (int j = 0; j < 8; ++j) {
            const int row = wm * 16 + (h ? j + 8 : j);
            const int col = wn * 64 + t * 16 + r;
            Ct[row * 128 + col] = acc[t][j];
        }
    }
    __syncthreads();

    for (int idx = tid; idx < 64 * 128; idx += 256) {
        const int row = idx >> 7;
        const int col = idx & 127;
        float x = Ct[row * 128 + col];
        if (mode == 1) {
            const float pos = (float)(r0 + row);
            const int   fi  = col & 63;
            const float ang = pos * __powf(10000.0f, -(float)(2 * fi) * (1.0f / 128.0f));
            const float c   = __cosf(ang);
            const float s   = __sinf(ang);
            const float rot = (col < 64) ? -Ct[row * 128 + col + 64]
                                         :  Ct[row * 128 + col - 64];
            x = x * c + rot * s;
            dst[(size_t)head * S_LEN * HD + (size_t)(r0 + row) * HD + col] = (__bf16)x;
        } else {
            dst[(size_t)head * HD * S_LEN + (size_t)col * S_LEN + (r0 + row)] = (__bf16)x;
        }
    }
}

// ---------------- Flash attention: one WG per (head, 128 queries) ------------------
__global__ __launch_bounds__(256)
void flash_attn_kernel(const __bf16* __restrict__ Qb, const __bf16* __restrict__ Kb,
                       const __bf16* __restrict__ Vt, __bf16* __restrict__ Ob) {
    __shared__ __align__(32) __bf16 Kt[32 * 128];   // [key][d]
    __shared__ __align__(32) __bf16 Vti[128 * 32];  // [d][key]
    __shared__ __align__(32) __bf16 Ps[8][16 * 32]; // per-wave P scratch

    const int tid  = threadIdx.x;
    const int wave = tid >> 5;
    const int lane = tid & 31;
    const int r    = lane & 15;
    const int h    = lane >> 4;
    const int head = blockIdx.y;
    const int qb0  = blockIdx.x * 128;
    const int q0   = qb0 + wave * 16;
    const int wave_qmax = q0 + 15;
    const float scale = 0.08838834764831845f; // 1/sqrt(128)

    const size_t hq = (size_t)head * S_LEN * HD;
    const size_t hv = (size_t)head * HD * S_LEN;

    v16bf qf[4];
#pragma unroll
    for (int c = 0; c < 4; ++c)
        qf[c] = load_a_frag(Qb + hq + (size_t)q0 * HD + c * 32, HD);

    v8f o[8] = {};
    float mrow[8], lrow[8];
#pragma unroll
    for (int j = 0; j < 8; ++j) { mrow[j] = -1e30f; lrow[j] = 0.0f; }

    // cooperative staging indices (256 threads x one b128 each per tile)
    const int skey = tid >> 3;            // 0..31
    const int sd0  = (tid & 7) * 16;      // 0..112
    const int sdd  = tid >> 1;            // 0..127
    const int skk  = (tid & 1) * 16;      // 0,16

    const int kmax = qb0 + 128;
    for (int kb0 = 0; kb0 < kmax; kb0 += 32) {
        // async global->LDS staging of K [key][d] and V^T [d][key] tiles (ASYNCcnt)
        async_copy_b128(Kb + hq + (size_t)(kb0 + skey) * HD + sd0, Kt + skey * 128 + sd0);
        async_copy_b128(Vt + hv + (size_t)sdd * S_LEN + kb0 + skk, Vti + sdd * 32 + skk);
        wait_asynccnt0();
        __syncthreads();

        if (kb0 <= wave_qmax) {   // wave-uniform: EXEC stays all-ones for WMMA
            v8f s0 = {}, s1 = {};
#pragma unroll
            for (int c = 0; c < 4; ++c) {
                v16bf b0 = load_b_frag(Kt + 0 * 128 + c * 32, 128);
                v16bf b1 = load_b_frag(Kt + 16 * 128 + c * 32, 128);
                s0 = wmma_bf16(qf[c], b0, s0);
                s1 = wmma_bf16(qf[c], b1, s1);
            }
#pragma unroll
            for (int j = 0; j < 8; ++j) {
                const int prow = (h ? j + 8 : j);
                const int qrow = q0 + prow;
                float v0 = s0[j] * scale;
                float v1 = s1[j] * scale;
                if (kb0 + r      > qrow) v0 = -1e30f;
                if (kb0 + 16 + r > qrow) v1 = -1e30f;
                float rm = fmaxf(v0, v1);
#pragma unroll
                for (int msk = 8; msk >= 1; msk >>= 1)
                    rm = fmaxf(rm, __shfl_xor(rm, msk, 16));
                const float nm    = fmaxf(mrow[j], rm);
                const float alpha = __expf(mrow[j] - nm);
                const float p0    = __expf(v0 - nm);
                const float p1    = __expf(v1 - nm);
                float ps = p0 + p1;
#pragma unroll
                for (int msk = 8; msk >= 1; msk >>= 1)
                    ps += __shfl_xor(ps, msk, 16);
                lrow[j] = lrow[j] * alpha + ps;
                mrow[j] = nm;
#pragma unroll
                for (int t = 0; t < 8; ++t) o[t][j] *= alpha;
                Ps[wave][prow * 32 + r]      = (__bf16)p0;
                Ps[wave][prow * 32 + 16 + r] = (__bf16)p1;
            }
            v16bf pf = load_a_frag(Ps[wave], 32);  // per-wave LDS: DS ops in-order
#pragma unroll
            for (int t = 0; t < 8; ++t) {
                v16bf vb = load_b_frag(Vti + (t * 16) * 32, 32);
                o[t] = wmma_bf16(pf, vb, o[t]);
            }
        }
        __syncthreads();
    }

    // normalize and write [s][head*128+d] bf16 for the output projection
#pragma unroll
    for (int j = 0; j < 8; ++j) {
        const int qrow = q0 + (h ? j + 8 : j);
        const float inv = 1.0f / lrow[j];
#pragma unroll
        for (int t = 0; t < 8; ++t)
            Ob[(size_t)qrow * HID + head * HD + t * 16 + r] = (__bf16)(o[t][j] * inv);
    }
}

// ---------------- Output projection: out = A(bf16) @ Wo^T, f32 out -----------------
__global__ __launch_bounds__(256)
void out_proj_kernel(const __bf16* __restrict__ A, const float* __restrict__ W,
                     float* __restrict__ out) {
    __shared__ __align__(32) __bf16 At[64 * 32];
    __shared__ __align__(32) __bf16 Bt[128 * 32];

    const int tid  = threadIdx.x;
    const int wave = tid >> 5;
    const int wm   = wave & 3;
    const int wn   = wave >> 2;
    const int lane = tid & 31;
    const int r    = lane & 15;
    const int h    = lane >> 4;
    const int r0   = blockIdx.x * 64;
    const int c0   = blockIdx.y * 128;

    v8f acc[4] = {};

    const int am = tid >> 2, ak = (tid & 3) * 8;
    const int bn = tid >> 1, bk = (tid & 1) * 16;

    for (int k0 = 0; k0 < HID; k0 += 32) {
        // A tile is already bf16: async copy straight into LDS (ASYNCcnt)
        async_copy_b128(A + (size_t)(r0 + am) * HID + k0 + ak, At + am * 32 + ak);
        {
            const float4* s = (const float4*)(W + (size_t)(c0 + bn) * HID + k0 + bk);
            __bf16* d = Bt + bn * 32 + bk;
#pragma unroll
            for (int q = 0; q < 4; ++q) {
                float4 x = s[q];
                d[q * 4 + 0] = (__bf16)x.x; d[q * 4 + 1] = (__bf16)x.y;
                d[q * 4 + 2] = (__bf16)x.z; d[q * 4 + 3] = (__bf16)x.w;
            }
        }
        if (k0 + 32 < HID) {
            __builtin_prefetch(A + (size_t)(r0 + am) * HID + k0 + 32 + ak, 0, 1);
            __builtin_prefetch(W + (size_t)(c0 + bn) * HID + k0 + 32 + bk, 0, 1);
        }
        wait_asynccnt0();
        __syncthreads();

        v16bf a = load_a_frag(At + (wm * 16) * 32, 32);
#pragma unroll
        for (int t = 0; t < 4; ++t) {
            v16bf b = load_b_frag(Bt + (wn * 64 + t * 16) * 32, 32);
            acc[t] = wmma_bf16(a, b, acc[t]);
        }
        __syncthreads();
    }

#pragma unroll
    for (int t = 0; t < 4; ++t)
#pragma unroll
        for (int j = 0; j < 8; ++j) {
            const int row = r0 + wm * 16 + (h ? j + 8 : j);
            const int col = c0 + wn * 64 + t * 16 + r;
            out[(size_t)row * HID + col] = acc[t][j];
        }
}

// ---------------------------------- launch -----------------------------------------
extern "C" void kernel_launch(void* const* d_in, const int* in_sizes, int n_in,
                              void* d_out, int out_size, void* d_ws, size_t ws_size,
                              hipStream_t stream) {
    const float* X  = (const float*)d_in[0];
    const float* Wq = (const float*)d_in[1];
    const float* Wk = (const float*)d_in[2];
    const float* Wv = (const float*)d_in[3];
    const float* Wo = (const float*)d_in[4];
    // d_in[5] = causal additive mask (applied analytically), d_in[6] = arange positions

    __bf16* Qb = (__bf16*)d_ws;                         // [NH][S][HD]
    __bf16* Kb = Qb + (size_t)NH * S_LEN * HD;          // [NH][S][HD]
    __bf16* Vt = Kb + (size_t)NH * S_LEN * HD;          // [NH][HD][S]
    __bf16* Ab = Vt + (size_t)NH * HD * S_LEN;          // [S][HID]

    const dim3 blk(256);
    qkv_proj_kernel<<<dim3(S_LEN / 64, NH), blk, 0, stream>>>(X, Wq, Qb, 1);
    qkv_proj_kernel<<<dim3(S_LEN / 64, NH), blk, 0, stream>>>(X, Wk, Kb, 1);
    qkv_proj_kernel<<<dim3(S_LEN / 64, NH), blk, 0, stream>>>(X, Wv, Vt, 0);
    flash_attn_kernel<<<dim3(S_LEN / 128, NH), blk, 0, stream>>>(Qb, Kb, Vt, Ab);
    out_proj_kernel<<<dim3(S_LEN / 64, HID / HD), blk, 0, stream>>>(Ab, Wo,
                                                                    (float*)d_out);
}
